// AdjLeafGNN_6854767805295
// MI455X (gfx1250) — matrix operand order repaired and checked by reading, not compiled, not run on hardware
//
#include <hip/hip_runtime.h>
#include <hip/hip_bf16.h>
#include <stdint.h>

typedef __attribute__((ext_vector_type(16))) _Float16 v16h;
typedef __attribute__((ext_vector_type(8)))  _Float16 v8h;
typedef __attribute__((ext_vector_type(8)))  float    v8f;

#define KNN 5
#define HID 256

// ---------------------------------------------------------------------------
// WMMA helpers (CDNA5 wave32, V_WMMA_F32_16X16X32_F16)
// ---------------------------------------------------------------------------
__device__ inline v8f wmma_f16(v16h a, v16h b, v8f c) {
  return __builtin_amdgcn_wmma_f32_16x16x32_f16(false, a, false, b, (short)0, c,
                                                false, false);
}

// A fragment from a row-chunk pointer: slot j<8 -> r[j], slot j>=8 -> r[16+j-8].
// (ISA 16-bit A layout: lane<16 holds K {0..7,16..23}; lane>=16 {8..15,24..31};
//  caller bakes half*8 into the pointer.)  Two aligned 16B loads.
__device__ inline v16h frag_a_ptr(const _Float16* r) {
  v8h lo = *(const v8h*)r;
  v8h hi = *(const v8h*)(r + 16);
  v16h a;
#pragma unroll
  for (int i = 0; i < 8; ++i) { a[i] = lo[i]; a[8 + i] = hi[i]; }
  return a;
}

// B fragment from fragment-native packed tiles [kb][n][32]: lane points at
// &P[kb][n][half*16] -> 16 contiguous halves (one aligned 32B load).
__device__ inline v16h frag_b_ptr(const _Float16* p) {
  return *(const v16h*)p;
}

// ---------------------------------------------------------------------------
// conv1: 3x3 s2 p1, 3->64 (~1% of FLOPs, VALU). Output spatial-major
// h1s[img][pos(256)][ch(64)] f16 so conv2 can stage LDS with b128 copies.
// ---------------------------------------------------------------------------
__global__ __launch_bounds__(256) void k_conv1(const float* __restrict__ x,
                                               const float* __restrict__ w,
                                               const float* __restrict__ b,
                                               _Float16* __restrict__ h1s) {
  int id = blockIdx.x * 256 + threadIdx.x;   // 1024*256*64
  int ch = id & 63;
  int p = (id >> 6) & 255;
  int img = id >> 14;
  int oy = p >> 4, ox = p & 15;
  float acc = b[ch];
  const float* xi = x + (size_t)img * 3072;
  const float* wc = w + ch * 27;
#pragma unroll
  for (int ci = 0; ci < 3; ++ci) {
#pragma unroll
    for (int ky = 0; ky < 3; ++ky) {
      int iy = oy * 2 + ky - 1;
      if ((unsigned)iy >= 32u) continue;
#pragma unroll
      for (int kx = 0; kx < 3; ++kx) {
        int ix = ox * 2 + kx - 1;
        if ((unsigned)ix >= 32u) continue;
        acc += xi[ci * 1024 + iy * 32 + ix] * wc[ci * 9 + kx + ky * 3];
      }
    }
  }
  h1s[(size_t)img * 16384 + p * 64 + ch] = (_Float16)fmaxf(acc, 0.f);
}

// ---------------------------------------------------------------------------
// Weight packers -> fragment-native B tiles P[kb][n(128 or N)][32] f16.
// Convs use tap-major K permutation: k' = tap*I + ci.
// ---------------------------------------------------------------------------
__global__ void k_prep_conv_bt(const float* __restrict__ W,   // [128][I][3][3]
                               _Float16* __restrict__ P, int I, int total) {
  int id = blockIdx.x * 256 + threadIdx.x;   // total = 9*I*128
  if (id >= total) return;
  int o = id & 127;
  int kp = id >> 7;
  int tap = kp / I, ci = kp % I;
  float v = W[(size_t)o * (9 * I) + ci * 9 + tap];
  P[(size_t)(kp >> 5) * 4096 + o * 32 + ((kp & 7) | ((kp & 8) << 1) | ((kp & 16) >> 1))] = (_Float16)v;
}
// ASPP branches 2,3 (dil 12,18 on 8x8 maps): only the center tap is in
// bounds -> exact 1x1 conv. Pack center-tap weights, K=128.
__global__ void k_prep_center(const float* __restrict__ W,    // [4][128][128][9]
                              _Float16* __restrict__ P) {
  int id = blockIdx.x * 256 + threadIdx.x;   // 2*128*128
  if (id >= 2 * 128 * 128) return;
  int o = id & 127, k = (id >> 7) & 127, brm = id >> 14;
  int br = brm + 2;
  float v = W[(size_t)((br * 128 + o) * 128 + k) * 9 + 4];
  P[(size_t)brm * 4 * 4096 + (size_t)(k >> 5) * 4096 + o * 32 + ((k & 7) | ((k & 8) << 1) | ((k & 16) >> 1))] =
      (_Float16)v;
}
// Plain matrix W[K][N] row-major -> tiles
__global__ void k_prep_bt(const float* __restrict__ W, _Float16* __restrict__ P,
                          int K, int N) {
  int id = blockIdx.x * 256 + threadIdx.x;
  if (id >= K * N) return;
  int k = id / N, n = id % N;
  P[(size_t)(k >> 5) * N * 32 + n * 32 + ((k & 7) | ((k & 8) << 1) | ((k & 16) >> 1))] = (_Float16)W[id];
}
// Transposed source W[N][K] row-major (1x1 conv weights [O][I]; emb for Gram)
__global__ void k_prep_bt_t(const float* __restrict__ W, _Float16* __restrict__ P,
                            int K, int N) {
  int id = blockIdx.x * 256 + threadIdx.x;
  if (id >= K * N) return;
  int n = id / K, k = id % K;
  P[(size_t)(k >> 5) * N * 32 + n * 32 + ((k & 7) | ((k & 8) << 1) | ((k & 16) >> 1))] = (_Float16)W[id];
}
__global__ void k_cast16(const float* __restrict__ in, _Float16* __restrict__ out,
                         int n) {
  int id = blockIdx.x * 256 + threadIdx.x;
  if (id < n) out[id] = (_Float16)in[id];
}

// ---------------------------------------------------------------------------
// conv2: implicit GEMM, per image M=64, N=128, K=576 (tap-major). Input image
// staged once in LDS as [pos(16x16)][ci(64)]; A frag = 2x ds_load_b128 with a
// branchless OOB select; B frags straight from packed global tiles.
// ---------------------------------------------------------------------------
__global__ __launch_bounds__(256) void k_conv2(const _Float16* __restrict__ h1s,
                                               const _Float16* __restrict__ wbt,
                                               const float* __restrict__ bias,
                                               _Float16* __restrict__ h2s) {
  __shared__ __align__(16) _Float16 sIn[256 * 64];   // 32 KB
  int img = blockIdx.x, tid = threadIdx.x;
  __builtin_prefetch(h1s + (size_t)(img + 1) * 16384, 0, 1);
  {
    const uint4* src = (const uint4*)(h1s + (size_t)img * 16384);
    uint4* dst = (uint4*)sIn;
#pragma unroll
    for (int u = 0; u < 8; ++u) dst[tid + 256 * u] = src[tid + 256 * u];
  }
  __syncthreads();
  int wave = tid >> 5, lane = tid & 31, half = lane >> 4, l = lane & 15;
  int mtile = wave & 3, ngroup = wave >> 2, m0 = mtile * 16;
  int m = m0 + l, oy = m >> 3, ox = m & 7;
  int hoff = half * 8;
  const _Float16* bb = wbt + (ngroup * 64 + l) * 32 + half * 16;
  v8f acc[4] = {};
#pragma unroll
  for (int kb = 0; kb < 18; ++kb) {
    const int tap = kb >> 1, ty = tap / 3, tx = tap % 3;
    const int cib = (kb & 1) * 32;
    int py = 2 * oy + (ty - 1), px = 2 * ox + (tx - 1);
    bool ok = ((unsigned)py < 16u) && ((unsigned)px < 16u);
    int off = (py * 16 + px) * 64 + cib + hoff;
    off = ok ? off : (1 << 22);              // LDS OOB loads return zero
    v16h a = frag_a_ptr(sIn + off);
#pragma unroll
    for (int nt = 0; nt < 4; ++nt) {
      v16h b = frag_b_ptr(bb + kb * 4096 + nt * 512);
      acc[nt] = wmma_f16(a, b, acc[nt]);
    }
  }
#pragma unroll
  for (int nt = 0; nt < 4; ++nt) {
    int ch = ngroup * 64 + nt * 16 + l;
    float bv = bias[ch];
#pragma unroll
    for (int r = 0; r < 8; ++r) {
      int pos = m0 + half * 8 + r;
      float v = fmaxf(acc[nt][r] + bv, 0.f);
      h2s[(size_t)img * 8192 + pos * 128 + ch] = (_Float16)v;   // [img][pos][ch]
    }
  }
}

// ---------------------------------------------------------------------------
// ASPP dilated 3x3 (branches 0,1: dil 1,6). Same scheme, K=1152 tap-major,
// input staged as [pos(64)][ci(128)]; DIL is a template constant.
// ---------------------------------------------------------------------------
template <int DIL>
__global__ __launch_bounds__(256) void k_aspp3(const _Float16* __restrict__ h2s,
                                               const _Float16* __restrict__ wbt,
                                               const float* __restrict__ bias,
                                               _Float16* __restrict__ catT,
                                               int br) {
  __shared__ __align__(16) _Float16 sIn[64 * 128];   // 16 KB
  int img = blockIdx.x, tid = threadIdx.x;
  __builtin_prefetch(h2s + (size_t)(img + 1) * 8192, 0, 1);
  {
    const uint4* src = (const uint4*)(h2s + (size_t)img * 8192);
    uint4* dst = (uint4*)sIn;
#pragma unroll
    for (int u = 0; u < 4; ++u) dst[tid + 256 * u] = src[tid + 256 * u];
  }
  __syncthreads();
  int wave = tid >> 5, lane = tid & 31, half = lane >> 4, l = lane & 15;
  int mtile = wave & 3, ngroup = wave >> 2, m0 = mtile * 16;
  int m = m0 + l, oy = m >> 3, ox = m & 7;
  int hoff = half * 8;
  const _Float16* bb = wbt + (ngroup * 64 + l) * 32 + half * 16;
  v8f acc[4] = {};
#pragma unroll
  for (int kb = 0; kb < 36; ++kb) {
    const int tap = kb >> 2, ty = tap / 3, tx = tap % 3;
    const int cib = (kb & 3) * 32;
    int py = oy + DIL * (ty - 1), px = ox + DIL * (tx - 1);
    bool ok = ((unsigned)py < 8u) && ((unsigned)px < 8u);
    int off = (py * 8 + px) * 128 + cib + hoff;
    off = ok ? off : (1 << 22);
    v16h a = frag_a_ptr(sIn + off);
#pragma unroll
    for (int nt = 0; nt < 4; ++nt) {
      v16h b = frag_b_ptr(bb + kb * 4096 + nt * 512);
      acc[nt] = wmma_f16(a, b, acc[nt]);
    }
  }
  const float* bbr = bias + br * 128;
#pragma unroll
  for (int nt = 0; nt < 4; ++nt) {
    int ch = ngroup * 64 + nt * 16 + l;
    float bv = bbr[ch];
#pragma unroll
    for (int r = 0; r < 8; ++r) {
      int pos = m0 + half * 8 + r;
      float v = fmaxf(acc[nt][r] + bv, 0.f);
      catT[(size_t)img * 32768 + pos * 512 + br * 128 + ch] = (_Float16)v;
    }
  }
}

// ASPP branches 2,3 as exact 1x1 convs: pure GEMM, no LDS, no barriers.
__global__ __launch_bounds__(256) void k_aspp11(const _Float16* __restrict__ h2s,
                                                const _Float16* __restrict__ wc,
                                                const float* __restrict__ bias,
                                                _Float16* __restrict__ catT) {
  int img = blockIdx.x, brm = blockIdx.y, br = brm + 2, tid = threadIdx.x;
  int wave = tid >> 5, lane = tid & 31, half = lane >> 4, l = lane & 15;
  int mtile = wave & 3, ngroup = wave >> 2, m0 = mtile * 16;
  const _Float16* arow = h2s + (size_t)img * 8192 + (m0 + l) * 128 + half * 8;
  const _Float16* bb =
      wc + (size_t)brm * 4 * 4096 + (ngroup * 64 + l) * 32 + half * 16;
  v8f acc[4] = {};
#pragma unroll
  for (int kb = 0; kb < 4; ++kb) {
    v16h a = frag_a_ptr(arow + kb * 32);
#pragma unroll
    for (int nt = 0; nt < 4; ++nt) {
      v16h b = frag_b_ptr(bb + kb * 4096 + nt * 512);
      acc[nt] = wmma_f16(a, b, acc[nt]);
    }
  }
  const float* bbr = bias + br * 128;
#pragma unroll
  for (int nt = 0; nt < 4; ++nt) {
    int ch = ngroup * 64 + nt * 16 + l;
    float bv = bbr[ch];
#pragma unroll
    for (int r = 0; r < 8; ++r) {
      int pos = m0 + half * 8 + r;
      float v = fmaxf(acc[nt][r] + bv, 0.f);
      catT[(size_t)img * 32768 + pos * 512 + br * 128 + ch] = (_Float16)v;
    }
  }
}

// proj 1x1 conv 512->128 per image: pure GEMM from catT rows (ld=512).
__global__ __launch_bounds__(256) void k_proj(const _Float16* __restrict__ catT,
                                              const _Float16* __restrict__ pbt,
                                              const float* __restrict__ bias,
                                              _Float16* __restrict__ projT) {
  int img = blockIdx.x, tid = threadIdx.x;
  int wave = tid >> 5, lane = tid & 31, half = lane >> 4, l = lane & 15;
  int mtile = wave & 3, ngroup = wave >> 2, m0 = mtile * 16;
  const _Float16* arow = catT + (size_t)img * 32768 + (m0 + l) * 512 + half * 8;
  const _Float16* bb = pbt + (ngroup * 64 + l) * 32 + half * 16;
  v8f acc[4] = {};
#pragma unroll 4
  for (int kb = 0; kb < 16; ++kb) {
    v16h a = frag_a_ptr(arow + kb * 32);
#pragma unroll
    for (int nt = 0; nt < 4; ++nt) {
      v16h b = frag_b_ptr(bb + kb * 4096 + nt * 512);
      acc[nt] = wmma_f16(a, b, acc[nt]);
    }
  }
#pragma unroll
  for (int nt = 0; nt < 4; ++nt) {
    int ch = ngroup * 64 + nt * 16 + l;
    float bv = bias[ch];
#pragma unroll
    for (int r = 0; r < 8; ++r) {
      int pos = m0 + half * 8 + r;
      float v = fmaxf(acc[nt][r] + bv, 0.f);
      projT[(size_t)img * 8192 + pos * 128 + ch] = (_Float16)v;
    }
  }
}

// Global average pool over 64 positions (coalesced: ch-innermost layout).
__global__ void k_gap(const _Float16* __restrict__ projT,
                      _Float16* __restrict__ gap) {
  int id = blockIdx.x * 256 + threadIdx.x;   // 1024*128
  int img = id >> 7, ch = id & 127;
  const _Float16* p = projT + (size_t)img * 8192 + ch;
  float s = 0.f;
#pragma unroll
  for (int i = 0; i < 64; ++i) s += (float)p[i * 128];
  gap[id] = (_Float16)(s * (1.f / 64.f));
}

// ---------------------------------------------------------------------------
// Generic WMMA GEMM, LDS-free: C[M][N] = A[M][K]f16 * Bpacked (+bias)(+relu).
// Block = 64x128 tile, 8 waves x (16x64). K%32==0, N%128==0.
// ---------------------------------------------------------------------------
__global__ __launch_bounds__(256) void k_gemm(const _Float16* __restrict__ A,
                                              const _Float16* __restrict__ Pb,
                                              float* __restrict__ C, int K, int N,
                                              const float* __restrict__ bias,
                                              int relu) {
  int tid = threadIdx.x;
  int mblk = blockIdx.x * 64, nblk = blockIdx.y * 128;
  int wave = tid >> 5, lane = tid & 31, half = lane >> 4, l = lane & 15;
  int mtile = wave & 3, ngroup = wave >> 2;
  const _Float16* arow = A + (size_t)(mblk + mtile * 16 + l) * K + half * 8;
  const _Float16* bb =
      Pb + (size_t)(nblk + ngroup * 64 + l) * 32 + half * 16;
  v8f acc[4] = {};
  int nkb = K >> 5;
#pragma unroll 4
  for (int kb = 0; kb < nkb; ++kb) {
    v16h a = frag_a_ptr(arow + kb * 32);
#pragma unroll
    for (int nt = 0; nt < 4; ++nt) {
      v16h b = frag_b_ptr(bb + (size_t)kb * N * 32 + nt * 512);
      acc[nt] = wmma_f16(a, b, acc[nt]);
    }
  }
#pragma unroll
  for (int nt = 0; nt < 4; ++nt) {
    int col = nblk + ngroup * 64 + nt * 16 + l;
    float bv = bias ? bias[col] : 0.f;
#pragma unroll
    for (int r = 0; r < 8; ++r) {
      int row = mblk + mtile * 16 + half * 8 + r;
      float v = acc[nt][r] + bv;
      if (relu) v = fmaxf(v, 0.f);
      C[(size_t)row * N + col] = v;
    }
  }
}

// ---------------------------------------------------------------------------
// kNN graph + GCN support kernels (bandwidth-trivial)
// ---------------------------------------------------------------------------
__global__ void k_sqrow(const float* __restrict__ emb, float* __restrict__ sq) {
  __shared__ float red[256];
  int i = blockIdx.x, t = threadIdx.x;
  float v = emb[(size_t)i * 256 + t];
  red[t] = v * v;
  __syncthreads();
  for (int s = 128; s > 0; s >>= 1) {
    if (t < s) red[t] += red[t + s];
    __syncthreads();
  }
  if (t == 0) sq[i] = red[0];
}

__global__ void k_dist(const float* __restrict__ gram, const float* __restrict__ sq,
                       float* __restrict__ dist) {
  int id = blockIdx.x * 256 + threadIdx.x;
  int i = id >> 10, j = id & 1023;
  float d2 = sq[i] + sq[j] - 2.f * gram[id];
  dist[id] = sqrtf(fmaxf(d2, 0.f));
}

__global__ void k_topk(const float* __restrict__ dist, int* __restrict__ idx) {
  int i = blockIdx.x * 256 + threadIdx.x;
  if (i >= 1024) return;
  float bd[KNN];
  int bi[KNN];
#pragma unroll
  for (int q = 0; q < KNN; ++q) { bd[q] = 3.4e38f; bi[q] = 0; }
  const float* row = dist + (size_t)i * 1024;
  for (int j = 0; j < 1024; ++j) {
    if (j == i) continue;
    float d = row[j];
    if (d < bd[KNN - 1]) {
      int q = KNN - 1;
      while (q > 0 && bd[q - 1] > d) {
        bd[q] = bd[q - 1]; bi[q] = bi[q - 1]; --q;
      }
      bd[q] = d; bi[q] = j;
    }
  }
#pragma unroll
  for (int q = 0; q < KNN; ++q) idx[i * KNN + q] = bi[q];
}

__global__ void k_adj(const int* __restrict__ idx, float* __restrict__ adj,
                      float* __restrict__ dinv) {
  __shared__ float red[256];
  int i = blockIdx.x, t = threadIdx.x;
  float* row = adj + (size_t)i * 1024;
#pragma unroll
  for (int u = 0; u < 4; ++u) row[t + 256 * u] = 0.f;
  __syncthreads();
  if (t < KNN) row[idx[i * KNN + t]] = 1.f;
  if (t == KNN) row[i] = 1.f;
  __syncthreads();
  float s = 0.f;
#pragma unroll
  for (int u = 0; u < 4; ++u) s += row[t + 256 * u];
  red[t] = s;
  __syncthreads();
  for (int st = 128; st > 0; st >>= 1) {
    if (t < st) red[t] += red[t + st];
    __syncthreads();
  }
  if (t == 0) dinv[i] = rsqrtf(fmaxf(red[0], 1e-12f));
}

__global__ void k_agg(const float* __restrict__ xw, const int* __restrict__ idx,
                      const float* __restrict__ dinv, const float* __restrict__ bias,
                      float* __restrict__ out, _Float16* __restrict__ out16,
                      int relu) {
  int i = blockIdx.x, t = threadIdx.x;
  float di = dinv[i];
  float s = di * xw[(size_t)i * HID + t];
#pragma unroll
  for (int q = 0; q < KNN; ++q) {
    int j = idx[i * KNN + q];
    s += dinv[j] * xw[(size_t)j * HID + t];
  }
  float v = di * s + bias[t];
  if (relu) v = fmaxf(v, 0.f);
  out[(size_t)i * HID + t] = v;
  if (out16) out16[(size_t)i * HID + t] = (_Float16)v;
}

__global__ void k_heads(const float* __restrict__ x, const float* __restrict__ cw,
                        const float* __restrict__ cb, const float* __restrict__ sw,
                        const float* __restrict__ sb, float* __restrict__ out_cls,
                        float* __restrict__ out_spr) {
  int id = blockIdx.x * 256 + threadIdx.x;
  if (id >= 1024 * 11) return;
  int i = id / 11, c = id % 11;
  const float* xi = x + (size_t)i * HID;
  if (c < 10) {
    float s = cb[c];
    for (int k = 0; k < HID; ++k) s += xi[k] * cw[k * 10 + c];
    out_cls[i * 10 + c] = s;
  } else {
    float s = sb[0];
    for (int k = 0; k < HID; ++k) s += xi[k] * sw[k];
    out_spr[i] = s;
  }
}

// ---------------------------------------------------------------------------
extern "C" void kernel_launch(void* const* d_in, const int* in_sizes, int n_in,
                              void* d_out, int out_size, void* d_ws, size_t ws_size,
                              hipStream_t stream) {
  (void)in_sizes; (void)n_in; (void)out_size; (void)ws_size;

  const float* images = (const float*)d_in[0];
  const float* w1     = (const float*)d_in[1];
  const float* b1     = (const float*)d_in[2];
  const float* w2     = (const float*)d_in[3];
  const float* b2     = (const float*)d_in[4];
  const float* aspp_w = (const float*)d_in[5];
  const float* aspp_b = (const float*)d_in[6];
  const float* proj_w = (const float*)d_in[7];
  const float* proj_b = (const float*)d_in[8];
  const float* fc_w   = (const float*)d_in[9];
  const float* fc_b   = (const float*)d_in[10];
  const float* gcn1_w = (const float*)d_in[11];
  const float* gcn1_b = (const float*)d_in[12];
  const float* gcn2_w = (const float*)d_in[13];
  const float* gcn2_b = (const float*)d_in[14];
  const float* cls_w  = (const float*)d_in[15];
  const float* cls_b  = (const float*)d_in[16];
  const float* spr_w  = (const float*)d_in[17];
  const float* spr_b  = (const float*)d_in[18];

  float* out_emb  = (float*)d_out;
  float* out_adj  = out_emb + 1024 * 256;
  float* out_dist = out_adj + 1024 * 1024;
  float* out_cls  = out_dist + 1024 * 1024;
  float* out_spr  = out_cls + 1024 * 10;

  char* ws = (char*)d_ws;
  size_t off = 0;
  auto alloc = [&](size_t bytes) -> void* {
    void* p = ws + off;
    off = (off + bytes + 255) & ~(size_t)255;
    return p;
  };
  _Float16* h1s    = (_Float16*)alloc((size_t)1024 * 16384 * 2);
  _Float16* h2s    = (_Float16*)alloc((size_t)1024 * 8192 * 2);
  _Float16* catT   = (_Float16*)alloc((size_t)1024 * 32768 * 2);
  _Float16* projT  = (_Float16*)alloc((size_t)1024 * 8192 * 2);
  _Float16* gapf16 = (_Float16*)alloc((size_t)1024 * 128 * 2);
  _Float16* w2g    = (_Float16*)alloc((size_t)18 * 4096 * 2);
  _Float16* asppg  = (_Float16*)alloc((size_t)2 * 36 * 4096 * 2);
  _Float16* asppc  = (_Float16*)alloc((size_t)2 * 4 * 4096 * 2);
  _Float16* projg  = (_Float16*)alloc((size_t)16 * 4096 * 2);
  _Float16* fcg    = (_Float16*)alloc((size_t)4 * 256 * 32 * 2);
  _Float16* gcn1g  = (_Float16*)alloc((size_t)8 * 256 * 32 * 2);
  _Float16* gcn2g  = (_Float16*)alloc((size_t)8 * 256 * 32 * 2);
  _Float16* embf16 = (_Float16*)alloc((size_t)1024 * 256 * 2);
  _Float16* embBT  = (_Float16*)alloc((size_t)8 * 1024 * 32 * 2);
  float*    gram   = (float*)alloc((size_t)1024 * 1024 * 4);
  float*    sq     = (float*)alloc((size_t)1024 * 4);
  int*      idx    = (int*)alloc((size_t)1024 * KNN * 4);
  float*    dinv   = (float*)alloc((size_t)1024 * 4);
  float*    xw1    = (float*)alloc((size_t)1024 * 256 * 4);
  float*    x1     = (float*)alloc((size_t)1024 * 256 * 4);
  _Float16* x1f16  = (_Float16*)alloc((size_t)1024 * 256 * 2);
  float*    xw2    = (float*)alloc((size_t)1024 * 256 * 4);
  float*    x2     = (float*)alloc((size_t)1024 * 256 * 4);

  // --- weight packing into fragment-native tiles ---
  k_prep_conv_bt<<<(576 * 128 + 255) / 256, 256, 0, stream>>>(w2, w2g, 64,
                                                              576 * 128);
  k_prep_conv_bt<<<(1152 * 128 + 255) / 256, 256, 0, stream>>>(
      aspp_w, asppg, 128, 1152 * 128);                         // branch 0 (dil 1)
  k_prep_conv_bt<<<(1152 * 128 + 255) / 256, 256, 0, stream>>>(
      aspp_w + (size_t)128 * 128 * 9, asppg + (size_t)36 * 4096, 128,
      1152 * 128);                                             // branch 1 (dil 6)
  k_prep_center<<<(2 * 128 * 128 + 255) / 256, 256, 0, stream>>>(aspp_w, asppc);
  k_prep_bt_t<<<(512 * 128 + 255) / 256, 256, 0, stream>>>(proj_w, projg, 512, 128);
  k_prep_bt<<<(128 * 256 + 255) / 256, 256, 0, stream>>>(fc_w, fcg, 128, 256);
  k_prep_bt<<<(256 * 256 + 255) / 256, 256, 0, stream>>>(gcn1_w, gcn1g, 256, 256);
  k_prep_bt<<<(256 * 256 + 255) / 256, 256, 0, stream>>>(gcn2_w, gcn2g, 256, 256);

  // --- encoder ---
  k_conv1<<<(1024 * 64 * 256) / 256, 256, 0, stream>>>(images, w1, b1, h1s);
  k_conv2<<<1024, 256, 0, stream>>>(h1s, w2g, b2, h2s);
  k_aspp3<1><<<1024, 256, 0, stream>>>(h2s, asppg, aspp_b, catT, 0);
  k_aspp3<6><<<1024, 256, 0, stream>>>(h2s, asppg + (size_t)36 * 4096, aspp_b,
                                       catT, 1);
  k_aspp11<<<dim3(1024, 2), 256, 0, stream>>>(h2s, asppc, aspp_b, catT);
  k_proj<<<1024, 256, 0, stream>>>(catT, projg, proj_b, projT);
  k_gap<<<(1024 * 128) / 256, 256, 0, stream>>>(projT, gapf16);
  // emb = gap @ fc_w + fc_b (M=1024,K=128,N=256)
  k_gemm<<<dim3(16, 2), 256, 0, stream>>>(gapf16, fcg, out_emb, 128, 256, fc_b, 0);

  // --- kNN graph ---
  k_cast16<<<(1024 * 256) / 256, 256, 0, stream>>>(out_emb, embf16, 1024 * 256);
  k_prep_bt_t<<<(1024 * 256 + 255) / 256, 256, 0, stream>>>(out_emb, embBT, 256,
                                                            1024);
  // gram = emb @ emb^T (M=1024,K=256,N=1024)
  k_gemm<<<dim3(16, 8), 256, 0, stream>>>(embf16, embBT, gram, 256, 1024, nullptr,
                                          0);
  k_sqrow<<<1024, 256, 0, stream>>>(out_emb, sq);
  k_dist<<<(1024 * 1024) / 256, 256, 0, stream>>>(gram, sq, out_dist);
  k_topk<<<4, 256, 0, stream>>>(out_dist, idx);
  k_adj<<<1024, 256, 0, stream>>>(idx, out_adj, dinv);

  // --- GCN layers ---
  k_gemm<<<dim3(16, 2), 256, 0, stream>>>(embf16, gcn1g, xw1, 256, 256, nullptr, 0);
  k_agg<<<1024, 256, 0, stream>>>(xw1, idx, dinv, gcn1_b, x1, x1f16, 1);
  k_gemm<<<dim3(16, 2), 256, 0, stream>>>(x1f16, gcn2g, xw2, 256, 256, nullptr, 0);
  k_agg<<<1024, 256, 0, stream>>>(xw2, idx, dinv, gcn2_b, x2, nullptr, 0);

  // --- heads ---
  k_heads<<<(1024 * 11 + 255) / 256, 256, 0, stream>>>(x2, cls_w, cls_b, spr_w,
                                                       spr_b, out_cls, out_spr);
}
